// Discriminator_61100204753392
// MI455X (gfx1250) — matrix operand order repaired and checked by reading, not compiled, not run on hardware
//
#include <hip/hip_runtime.h>

// ---------------------------------------------------------------------------
// Fused 2-layer GRU (Keras reset_after=True) for gfx1250 (CDNA5).
//   - emb@W1 folded into a 25x300 lookup table P1 (setup kernel).
//   - U1/W2/U2 pre-swizzled into bf16 WMMA B-fragments (setup kernel -> ws),
//     block-copied into LDS by the persistent kernel.
//   - One workgroup (128 threads = 4 wave32) owns 16 batch rows and runs the
//     full T=512 scan of BOTH layers with v_wmma_f32_16x16x32_bf16.
//   - Gate pass maintains h both as f32 and as bf16 A-fragments in LDS so the
//     GEMMs load A with ds_load_b128 instead of scalar gathers + converts.
// ---------------------------------------------------------------------------

typedef __attribute__((ext_vector_type(16))) __bf16 v16bf;
typedef __attribute__((ext_vector_type(8)))  float  v8f;

#define B_   512
#define T_   512
#define E_   128
#define U_   100
#define TU_  300
#define NT_  19     // N tiles of 16 covering 304 (300 padded)
#define KC_  4      // K chunks of 32 covering 128 (100 padded)
#define P1LD 304    // row stride of the P1 lookup table
#define HLD  104    // row stride of f32 h state in LDS
#define RLD  304    // row stride of rec/x2 buffers in LDS

// --- LDS layout (bytes) -----------------------------------------------------
#define WFRAG_HALFS (3 * KC_ * NT_ * 512)        // 116736 bf16 = 233472 B
#define OFF_BUFA    (WFRAG_HALFS * 2)
#define OFF_BUFB    (OFF_BUFA + 16 * RLD * 4)
#define OFF_H1      (OFF_BUFB + 16 * RLD * 4)
#define OFF_H2      (OFF_H1   + 16 * HLD * 4)
#define OFF_BR1     (OFF_H2   + 16 * HLD * 4)
#define OFF_B20     (OFF_BR1  + RLD * 4)
#define OFF_B21     (OFF_B20  + RLD * 4)
#define OFF_TOK     (OFF_B21  + RLD * 4)
#define OFF_A1F     (OFF_TOK + 64)               // h1 A-fragments: 4*512 bf16
#define OFF_A2F     (OFF_A1F + KC_ * 512 * 2)    // h2 A-fragments
#define LDS_BYTES   (OFF_A2F + KC_ * 512 * 2)    // 297,600 B  (< 320 KB/WGP)

// --- workspace layout (bytes) ----------------------------------------------
#define WS_P1_BYTES   (25 * P1LD * 4)            // 30,400
#define WS_FRAG_OFF   30464                      // 64B-aligned
#define WS_FRAG_BYTES (WFRAG_HALFS * 2)          // 233,472

__device__ __forceinline__ unsigned short f2bf(float x) {
  union { float f; unsigned int i; } u; u.f = x;
  return (unsigned short)((u.i + 0x7FFFu + ((u.i >> 16) & 1u)) >> 16);
}
// fast gates: v_exp_f32 + v_rcp_f32 (no IEEE div fixup on the serial path)
__device__ __forceinline__ float sigmoidf_(float x) {
  return __builtin_amdgcn_rcpf(1.0f + __expf(-x));
}
__device__ __forceinline__ float tanhf_(float x) {
  return 2.0f * __builtin_amdgcn_rcpf(1.0f + __expf(-2.0f * x)) - 1.0f;
}

// A-fragment slot for h[m][k] (ISA 16-bit A layout, K = 32c + 16g + 8hs + e7)
__device__ __forceinline__ int afrag_slot(int m, int k) {
  const int c  = k >> 5, k32 = k & 31;
  const int hs = (k32 >> 3) & 1;
  const int e  = ((k32 >> 4) << 3) | (k32 & 7);
  return ((c << 5) + (m | (hs << 4))) * 16 + e;
}

// One wave computes its share (ntiles wn, wn+4, ...) of out16x304 = h @ Wg.
__device__ __forceinline__ void gemm16(const unsigned short* __restrict__ wfrag,
                                       int g,
                                       const unsigned short* __restrict__ afrag,
                                       float* __restrict__ obuf,
                                       int lane, int wn) {
  const int Mrow = lane & 15;
  const int hs   = (lane >> 4) & 1;
  v16bf a[KC_];
#pragma unroll
  for (int c = 0; c < KC_; ++c)
    a[c] = *(const v16bf*)(afrag + ((c << 5) + lane) * 16);   // 2x ds_load_b128

  for (int n = wn; n < NT_; n += 4) {             // scalar loop, EXEC all 1s
    v8f acc = {0.f, 0.f, 0.f, 0.f, 0.f, 0.f, 0.f, 0.f};
#pragma unroll
    for (int c = 0; c < KC_; ++c) {
      const v16bf bfr =
          *(const v16bf*)(wfrag + (((size_t)(g * KC_ + c) * NT_ + n) * 32 + lane) * 16);
      acc = __builtin_amdgcn_wmma_f32_16x16x32_bf16(
          false, a[c], false, bfr, (short)0, acc, false, false);
    }
    const int col = n * 16 + Mrow;                // C/D layout: M = r + 8*hs
#pragma unroll
    for (int r = 0; r < 8; ++r)
      obuf[(r + 8 * hs) * RLD + col] = acc[r];
  }
}

// --------------------------- setup: P1 = emb@W1 + b1[0] ---------------------
__global__ void p1_setup(const float* __restrict__ emb,
                         const float* __restrict__ W1,
                         const float* __restrict__ b1,
                         float* __restrict__ P1) {
  const int v = blockIdx.x;
  const int k = threadIdx.x;
  if (k >= P1LD) return;
  float acc = 0.0f;
  if (k < TU_) {
    acc = b1[k];
    for (int e = 0; e < E_; ++e) acc += emb[v * E_ + e] * W1[e * TU_ + k];
  }
  P1[v * P1LD + k] = acc;
}

// ------------- setup: swizzle U1/W2/U2 into bf16 WMMA B-fragments -----------
__global__ void wswz_setup(const float* __restrict__ U1,
                           const float* __restrict__ W2,
                           const float* __restrict__ U2,
                           unsigned short* __restrict__ gfrag) {
  const int idx = blockIdx.x * blockDim.x + threadIdx.x;
  if (idx >= WFRAG_HALFS) return;
  const int fi = idx >> 9, le = idx & 511;
  const int l = le >> 4, e = le & 15;
  const int g = fi / (KC_ * NT_);
  const int rc = fi - g * (KC_ * NT_);
  const int c = rc / NT_, n = rc - c * NT_;
  const int N = n * 16 + (l & 15);                // B layout: N = lane&15,
  const int K = 32 * c + ((e >> 3) << 4) + ((l >> 4) << 3) + (e & 7);  // K = 16*(e>>3)+8*(lane>>4)+(e&7)
  const float* src = (g == 0) ? U1 : (g == 1) ? W2 : U2;
  const float w = (K < U_ && N < TU_) ? src[K * TU_ + N] : 0.0f;
  gfrag[idx] = f2bf(w);
}

// --------------------------- main fused GRU scan ----------------------------
__global__ __launch_bounds__(128) void gru_fused(
    const int*   __restrict__ tokens,
    const float* __restrict__ b1, const float* __restrict__ b2,
    const float* __restrict__ Wd, const float* __restrict__ bd,
    const float* __restrict__ P1,
    const unsigned short* __restrict__ gfrag,
    float* __restrict__ out) {
  extern __shared__ char smem[];
  unsigned short* wfrag  = (unsigned short*)smem;
  float* bufA = (float*)(smem + OFF_BUFA);
  float* bufB = (float*)(smem + OFF_BUFB);
  float* h1   = (float*)(smem + OFF_H1);
  float* h2   = (float*)(smem + OFF_H2);
  float* br1  = (float*)(smem + OFF_BR1);
  float* b20  = (float*)(smem + OFF_B20);
  float* b21  = (float*)(smem + OFF_B21);
  int*   stok = (int*)  (smem + OFF_TOK);
  unsigned short* a1f = (unsigned short*)(smem + OFF_A1F);
  unsigned short* a2f = (unsigned short*)(smem + OFF_A2F);

  const int tid  = threadIdx.x;
  const int lane = tid & 31;
  const int wn   = __builtin_amdgcn_readfirstlane(tid >> 5);  // uniform wave id
  const int m0   = blockIdx.x * 16;

  // ---- bulk copy pre-swizzled weight fragments global -> LDS (b128 ops)
  {
    const uint4* src = (const uint4*)gfrag;
    uint4* dst = (uint4*)wfrag;
    for (int i = tid; i < WFRAG_HALFS / 8; i += 128) dst[i] = src[i];
  }
  for (int k = tid; k < RLD; k += 128) {
    br1[k] = (k < TU_) ? b1[TU_ + k] : 0.f;   // recurrent bias, layer 1
    b20[k] = (k < TU_) ? b2[k]       : 0.f;   // input bias,     layer 2
    b21[k] = (k < TU_) ? b2[TU_ + k] : 0.f;   // recurrent bias, layer 2
  }
  for (int k = tid; k < 16 * HLD; k += 128) { h1[k] = 0.f; h2[k] = 0.f; }
  for (int k = tid; k < KC_ * 512; k += 128) { a1f[k] = 0; a2f[k] = 0; }
  __syncthreads();

  for (int t = 0; t < T_; ++t) {
    if (tid < 16) stok[tid] = tokens[(m0 + tid) * T_ + t];

    // rec1 = h1 @ U1  -> bufA
    gemm16(wfrag, 0, a1f, bufA, lane, wn);
    __syncthreads();

    // layer-1 gates (x-projection is the P1 lookup); update f32 h and A-frags
    for (int idx = tid; idx < 16 * U_; idx += 128) {
      const int m = idx / U_, k = idx - m * U_;
      const int tok = stok[m];
      const float* p = P1 + tok * P1LD;
      const float h  = h1[m * HLD + k];
      const float z  = sigmoidf_(p[k]      + bufA[m * RLD + k]      + br1[k]);
      const float r  = sigmoidf_(p[U_ + k] + bufA[m * RLD + U_ + k] + br1[U_ + k]);
      const float hh = tanhf_(p[2 * U_ + k] +
                              r * (bufA[m * RLD + 2 * U_ + k] + br1[2 * U_ + k]));
      float hn = z * h + (1.0f - z) * hh;
      hn = (tok != 0) ? hn : h;
      h1[m * HLD + k] = hn;
      a1f[afrag_slot(m, k)] = f2bf(hn);
    }
    __syncthreads();

    // x2 = h1 @ W2 -> bufA ;  rec2 = h2 @ U2 -> bufB
    gemm16(wfrag, 1, a1f, bufA, lane, wn);
    gemm16(wfrag, 2, a2f, bufB, lane, wn);
    __syncthreads();

    // layer-2 gates
    for (int idx = tid; idx < 16 * U_; idx += 128) {
      const int m = idx / U_, k = idx - m * U_;
      const int tok = stok[m];
      const float h  = h2[m * HLD + k];
      const float z  = sigmoidf_(bufA[m * RLD + k]      + b20[k] +
                                 bufB[m * RLD + k]      + b21[k]);
      const float r  = sigmoidf_(bufA[m * RLD + U_ + k] + b20[U_ + k] +
                                 bufB[m * RLD + U_ + k] + b21[U_ + k]);
      const float hh = tanhf_(bufA[m * RLD + 2 * U_ + k] + b20[2 * U_ + k] +
                              r * (bufB[m * RLD + 2 * U_ + k] + b21[2 * U_ + k]));
      float hn = z * h + (1.0f - z) * hh;
      hn = (tok != 0) ? hn : h;
      h2[m * HLD + k] = hn;
      a2f[afrag_slot(m, k)] = f2bf(hn);
    }
    __syncthreads();
  }

  // final dense: out[b] = h2 @ Wd + bd
  if (tid < 16) {
    float acc = bd[0];
    for (int k = 0; k < U_; ++k) acc += h2[tid * HLD + k] * Wd[k];
    out[m0 + tid] = acc;
  }
}

// ---------------------------------------------------------------------------
extern "C" void kernel_launch(void* const* d_in, const int* in_sizes, int n_in,
                              void* d_out, int out_size, void* d_ws, size_t ws_size,
                              hipStream_t stream) {
  const int*   tokens = (const int*)  d_in[0];
  const float* emb    = (const float*)d_in[1];
  const float* W1     = (const float*)d_in[2];
  const float* U1     = (const float*)d_in[3];
  const float* b1     = (const float*)d_in[4];
  const float* W2     = (const float*)d_in[5];
  const float* U2     = (const float*)d_in[6];
  const float* b2     = (const float*)d_in[7];
  const float* Wd     = (const float*)d_in[8];
  const float* bd     = (const float*)d_in[9];
  float* out = (float*)d_out;

  float*          P1    = (float*)d_ws;
  unsigned short* gfrag = (unsigned short*)((char*)d_ws + WS_FRAG_OFF);

  p1_setup<<<25, P1LD, 0, stream>>>(emb, W1, b1, P1);
  wswz_setup<<<(WFRAG_HALFS + 255) / 256, 256, 0, stream>>>(U1, W2, U2, gfrag);
  gru_fused<<<B_ / 16, 128, LDS_BYTES, stream>>>(
      tokens, b1, b2, Wd, bd, P1, gfrag, out);
}
